// CriticNetwork_30777735643697
// MI455X (gfx1250) — matrix operand order repaired and checked by reading, not compiled, not run on hardware
//
#include <hip/hip_runtime.h>

typedef __attribute__((ext_vector_type(16))) _Float16 v16h;
typedef __attribute__((ext_vector_type(8)))  float    v8f;

#define NB   128
#define NA   32      // agents N
#define SS   128     // state dim
#define AA   16      // action dim
#define DW   64      // key/query dim
#define H1   64      // hidden
#define W1K  272     // 2S+A
#define NN   1024    // N*N

__global__ __launch_bounds__(256) void critic_kernel(
    const float* __restrict__ g_states,   // [B,N,S]
    const float* __restrict__ g_policies, // [B,N,A]
    const float* __restrict__ g_actions,  // [B,N,A]
    const float* __restrict__ g_noise,    // [B,N,N,A]
    const float* __restrict__ g_Wk,       // [DW,S]
    const float* __restrict__ g_Wq,       // [DW,S]
    const float* __restrict__ g_W1,       // [H1,2S+A]
    const float* __restrict__ g_b1,       // [H1]
    const float* __restrict__ g_W2,       // [1,H1]
    const float* __restrict__ g_b2,       // [1]
    float* __restrict__ g_out)            // [B*N*N value | B*N*N weight]
{
  const int b    = blockIdx.x;
  const int tid  = threadIdx.x;
  const int lane = tid & 31;
  const int wave = tid >> 5;

  // ---- LDS ----
  __shared__ float s_noise[NN * AA];                // 64KB, async-DMA'd noise slab
  __shared__ unsigned char s_mix[24576];            // union region
  _Float16* s_states_h = (_Float16*)s_mix;          // [32][128] f16 (load..phaseA)
  float*    s_kz       = (float*)(s_mix + 8192);    // [32][64]   (phaseA..scores)
  float*    s_qz       = (float*)(s_mix + 16384);   // [32][64]
  __shared__ float    s_g1[NA * H1];        // states @ W1a^T  [j][d]
  __shared__ float    s_w[NN];              // scores -> softmax weights
  __shared__ float    s_act[NA * AA];
  __shared__ float    s_pol[NA * AA];
  __shared__ float    s_sz[NA * AA];        // sum_k of z-part  [i][a]
  __shared__ _Float16 s_w1c[H1 * AA];       // W1[:,256:272] f16 [d][a]
  __shared__ float    s_colsum[SS];
  __shared__ float    s_c0[H1];             // b1 + W1b @ colsum/32
  __shared__ float    s_w2[H1];

  // ============ kick off async DMA of this batch's noise into LDS ============
  // global_load_async_to_lds_b128: per-lane 16B  GVS: saddr(SGPR64)+vaddr(VGPR32)
  {
    const unsigned long long nbase = (unsigned long long)(g_noise + (size_t)b * (NN * AA));
    const unsigned int ldsbase = (unsigned int)(size_t)(&s_noise[0]);
    for (int k = 0; k < 16; ++k) {
      unsigned int off = (unsigned int)(tid + k * 256) * 16u;  // byte offset
      unsigned int lds = ldsbase + off;
      asm volatile("global_load_async_to_lds_b128 %0, %1, %2"
                   :: "v"(lds), "v"(off), "s"(nbase) : "memory");
    }
  }

  const float* stb = g_states + (size_t)b * (NA * SS);

  // ================= load stage =================
  for (int idx = tid; idx < NA * SS; idx += 256)
    s_states_h[idx] = (_Float16)stb[idx];
  for (int idx = tid; idx < NA * AA; idx += 256) {
    s_act[idx] = g_actions[(size_t)b * NA * AA + idx];
    s_pol[idx] = g_policies[(size_t)b * NA * AA + idx];
  }
  if (tid < SS) {                       // column sums of states
    float s = 0.f;
    for (int i = 0; i < NA; ++i) s += stb[i * SS + tid];
    s_colsum[tid] = s;
  }
  if (tid >= 128 && tid < 128 + H1) {   // W2 + W1c staging
    int d = tid - 128;
    s_w2[d] = g_W2[d];
    for (int a = 0; a < AA; ++a)
      s_w1c[d * AA + a] = (_Float16)g_W1[d * W1K + 2 * SS + a];
  }
  __syncthreads();

  // ======== phase A: Kz, Qz, G1 via v_wmma_f32_16x16x32_f16 ========
  // 24 tile jobs = 3 GEMMs x (2 m-tiles x 4 n-tiles); exactly 3 per wave.
  for (int r = 0; r < 3; ++r) {
    int job   = wave * 3 + r;
    int which = job >> 3;                       // 0=Kz 1=Qz 2=G1
    int tile  = job & 7;
    int mbase = (tile >> 2) * 16;
    int nbase = (tile & 3) * 16;
    const float* Wg = (which == 0) ? g_Wk : (which == 1) ? g_Wq : g_W1;
    int   wstride   = (which == 2) ? W1K : SS;
    float* Dst      = (which == 0) ? s_kz : (which == 1) ? s_qz : s_g1;

    const int m     = mbase + (lane & 15);
    const int koff  = (lane < 16) ? 0 : 8;      // A-frag half-wave K offset
    const int n     = nbase + (lane & 15);
    const int kkoff = (lane < 16) ? 0 : 16;     // unused; B-frag also uses koff

    v8f acc = {0.f, 0.f, 0.f, 0.f, 0.f, 0.f, 0.f, 0.f};
    for (int kc = 0; kc < 4; ++kc) {
      int kb = kc * 32;
      v16h a, bb;
      for (int c = 0; c < 8; ++c) {             // ISA 16-bit A layout
        a[c]     = s_states_h[m * SS + kb + koff + c];
        a[8 + c] = s_states_h[m * SS + kb + 16 + koff + c];
      }
      for (int c = 0; c < 16; ++c)              // B[k][n] = W[n][k]
        bb[c] = (_Float16)Wg[n * wstride + kb + ((c < 8) ? (koff + c) : (16 + koff + c - 8))];
      acc = __builtin_amdgcn_wmma_f32_16x16x32_f16(
              false, a, false, bb, (short)0, acc, false, false);
    }
    const int mrow = (lane < 16) ? 0 : 8;
    for (int v = 0; v < 8; ++v)
      Dst[(mbase + mrow + v) * DW + n] = acc[v];
  }
  __syncthreads();

  // ======== scores[i][j] = Kz[i].Qz[j] / sqrt(DW) ========
  for (int k = 0; k < 4; ++k) {
    int e = tid + k * 256;
    int i = e >> 5, j = e & 31;
    float s = 0.f;
    for (int d = 0; d < DW; ++d) s += s_kz[i * DW + d] * s_qz[j * DW + d];
    s_w[e] = s * 0.125f;
  }
  __syncthreads();

  // ======== softmax rows (t<32) | c0 vector (64<=t<128) ========
  if (tid < NA) {
    int i = tid;
    float mx = s_w[i * NA];
    for (int j = 1; j < NA; ++j) mx = fmaxf(mx, s_w[i * NA + j]);
    float sum = 0.f;
    for (int j = 0; j < NA; ++j) {
      float e = __expf(s_w[i * NA + j] - mx);
      s_w[i * NA + j] = e;
      sum += e;
    }
    float inv = 1.f / sum;
    for (int j = 0; j < NA; ++j) s_w[i * NA + j] *= inv;
  } else if (tid >= 64 && tid < 64 + H1) {
    int d = tid - 64;
    float acc = 0.f;
    for (int s = 0; s < SS; ++s)
      acc += g_W1[d * W1K + SS + s] * s_colsum[s];
    s_c0[d] = g_b1[d] + acc * (1.f / 32.f);
  }
  // noise DMA must be resident before the sz phase below
  asm volatile("s_wait_asynccnt 0x0" ::: "memory");
  __syncthreads();

  // ======== sz[i][a] = sum_j (w[i][j]*act[j][a] + noise[i][j][a])  (LDS) ======
  for (int k = 0; k < 2; ++k) {
    int e = tid + k * 256;                  // 512 entries
    int i = e >> 4, a = e & 15;
    float sn = 0.f, wa = 0.f;
    for (int j = 0; j < NA; ++j) {
      sn += s_noise[i * (NA * AA) + j * AA + a];
      wa += s_w[i * NA + j] * s_act[j * AA + a];
    }
    s_sz[e] = sn + wa;
  }
  // coalesced weight_z output
  for (int k = 0; k < 4; ++k) {
    int e = tid + k * 256;
    g_out[(size_t)NB * NN + (size_t)b * NN + e] = s_w[e];
  }
  __syncthreads();

  // ======== phase D: h = lrelu(nfz @ W1c^T + G1[j] + c0); value = h.W2 + b2 ====
  // nfz is computed on the fly from the LDS noise slab while building A-frags.
  const float b2v = g_b2[0];
  for (int r = 0; r < 8; ++r) {
    int t     = wave + r * 8;               // 64 M-tiles of 16 positions
    int pbase = t * 16;                     // = i*32 + jbase
    int jb    = (t & 1) * 16;
    const int i_t  = pbase >> 5;            // agent i (constant in tile)
    const int row  = pbase + (lane & 15);   // position p for this lane's A-row
    const int jrow = row & 31;              // agent j for this row
    const int koff = (lane < 16) ? 0 : 8;
    const int mrow = (lane < 16) ? 0 : 8;

    const float wz = s_w[i_t * NA + jrow];
    v16h a;                                 // K=0..15 real, 16..31 zero-padded
    for (int c = 0; c < 8; ++c) {
      int aa = koff + c;
      float z  = wz * s_act[jrow * AA + aa] + s_noise[row * AA + aa];
      float nf = (s_sz[i_t * AA + aa] - z + s_pol[jrow * AA + aa]) * (1.f / 32.f);
      a[c]     = (_Float16)nf;
      a[8 + c] = (_Float16)0.f;
    }
    float partial[8] = {0.f, 0.f, 0.f, 0.f, 0.f, 0.f, 0.f, 0.f};

    for (int nt = 0; nt < 4; ++nt) {
      int d = nt * 16 + (lane & 15);
      v16h bb;                              // B[k][n]=W1c[d][k] for k<16 else 0
      for (int c = 0; c < 16; ++c)
        bb[c] = (c < 8) ? s_w1c[d * AA + koff + c] : (_Float16)0.f;

      v8f cc;                               // preload bias: G1[j][d] + c0[d]
      for (int v = 0; v < 8; ++v)
        cc[v] = s_g1[(jb + mrow + v) * DW + d] + s_c0[d];

      v8f h = __builtin_amdgcn_wmma_f32_16x16x32_f16(
                false, a, false, bb, (short)0, cc, false, false);

      float w2d = s_w2[d];
      for (int v = 0; v < 8; ++v) {
        float x = h[v];
        x = (x > 0.f) ? x : 0.01f * x;      // leaky_relu(0.01)
        partial[v] += x * w2d;
      }
    }
    // reduce over the 16 lanes of each half-wave (16 d-columns)
    for (int mask = 1; mask < 16; mask <<= 1)
      for (int v = 0; v < 8; ++v)
        partial[v] += __shfl_xor(partial[v], mask, 32);
    if ((lane & 15) == 0) {
      for (int v = 0; v < 8; ++v)
        g_out[(size_t)b * NN + pbase + mrow + v] = b2v + partial[v];
    }
  }
}

extern "C" void kernel_launch(void* const* d_in, const int* in_sizes, int n_in,
                              void* d_out, int out_size, void* d_ws, size_t ws_size,
                              hipStream_t stream) {
  critic_kernel<<<dim3(NB), dim3(256), 0, stream>>>(
      (const float*)d_in[0], (const float*)d_in[1], (const float*)d_in[2],
      (const float*)d_in[3], (const float*)d_in[4], (const float*)d_in[5],
      (const float*)d_in[6], (const float*)d_in[7], (const float*)d_in[8],
      (const float*)d_in[9], (float*)d_out);
}
